// RWKV_Tmix_x070_44830868636014
// MI455X (gfx1250) — matrix operand order repaired and checked by reading, not compiled, not run on hardware
//
#include <hip/hip_runtime.h>
#include <hip/hip_bf16.h>
#include <math.h>

#define T_LEN 2048
#define C_DIM 1024
#define H_N   16
#define HS_   64
#define EPS_GN 0.00064f

typedef float  v8f     __attribute__((ext_vector_type(8)));
typedef __bf16 bf16x16 __attribute__((ext_vector_type(16)));

__device__ __forceinline__ unsigned short f2bf(float f) {
    unsigned int u = __float_as_uint(f);
    u += 0x7FFFu + ((u >> 16) & 1u);            // round-to-nearest-even
    return (unsigned short)(u >> 16);
}
__device__ __forceinline__ float sigmoidf_(float x) { return 1.f / (1.f + __expf(-x)); }

// ---------------------------------------------------------------------------
// One-shot weight canonicalization: dst[N,K] bf16 from fp32 src.
//   srcIsNK=1: src already [N,K] (Wr/Wk/Wv/Wo);  srcIsNK=0: src is [K,N].
// ---------------------------------------------------------------------------
__global__ void __launch_bounds__(256)
convert_weight_kernel(const float* __restrict__ src, unsigned short* __restrict__ dst,
                      int K, int N, int srcIsNK)
{
    size_t i = (size_t)blockIdx.x * 256 + threadIdx.x;
    size_t total = (size_t)N * K;
    if (i >= total) return;
    if (srcIsNK) {
        dst[i] = f2bf(src[i]);
    } else {
        size_t n = i / (size_t)K, k = i - n * (size_t)K;
        dst[i] = f2bf(src[k * (size_t)N + n]);
    }
}

// ---------------------------------------------------------------------------
// Token shift + six lerp mixes, emitted as bf16 GEMM inputs.
// ---------------------------------------------------------------------------
__global__ void __launch_bounds__(256)
mix_kernel(const float* __restrict__ x, const float* __restrict__ x_prev,
           const float* __restrict__ cr, const float* __restrict__ cw,
           const float* __restrict__ ck, const float* __restrict__ cv,
           const float* __restrict__ ca, const float* __restrict__ cg,
           unsigned short* __restrict__ xr, unsigned short* __restrict__ xw,
           unsigned short* __restrict__ xk, unsigned short* __restrict__ xv,
           unsigned short* __restrict__ xa, unsigned short* __restrict__ xg)
{
    int t = blockIdx.x;
    int c0 = threadIdx.x * 4;
    #pragma unroll
    for (int q = 0; q < 4; q++) {
        int c = c0 + q;
        size_t idx = (size_t)t * C_DIM + c;
        float xc   = x[idx];
        float prev = (t == 0) ? x_prev[c] : x[idx - C_DIM];
        float dx   = prev - xc;
        xr[idx] = f2bf(xc + dx * cr[c]);
        xw[idx] = f2bf(xc + dx * cw[c]);
        xk[idx] = f2bf(xc + dx * ck[c]);
        xv[idx] = f2bf(xc + dx * cv[c]);
        xa[idx] = f2bf(xc + dx * ca[c]);
        xg[idx] = f2bf(xc + dx * cg[c]);
    }
}

// ---------------------------------------------------------------------------
// bf16 WMMA GEMM: out[M,N] = act( A[M,K] @ B^T ), A bf16 [M,K], B bf16 [N,K].
// Template: BM x BN block tile, 4 waves in 2x2, per-wave tile TM x TN (16-units).
// K stepped 64 (2 x 32-chunks). Double-buffered LDS staging via
// global_load_async_to_lds_b128 (ASYNCcnt): issue stage s+1, wait stage s,
// barrier, TM*TN WMMAs per chunk.
// LDS rows are 40 halfs (80 B = 5x16 B) so every 16 B fragment read is aligned.
// Fragment gathers follow gfx1250 WMMA VGPR layouts (ISA 7.12.2):
//   A lane L: M=L&15, halfs = K[koff..koff+7], K[16+koff..+7], koff=(L>>4)*8
//   B lane L: N=L&15, halfs = K[koff..koff+7], K[16+koff..+7], koff=(L>>4)*8
//   C lane L: N=L&15, float j -> M = 8*(L>>4)+j
// ---------------------------------------------------------------------------
template<int BM, int BN, int TM, int TN>
__global__ void __launch_bounds__(128)
gemm_bf16_wmma(const unsigned short* __restrict__ A,
               const unsigned short* __restrict__ Bw,
               int M, int N, int K, int act,
               float* __restrict__ outF, unsigned short* __restrict__ outB)
{
    const int LDT = 40;
    __shared__ unsigned short As[2][2][BM * LDT];   // [buf][chunk][row*LDT+k]
    __shared__ unsigned short Bs[2][2][BN * LDT];

    const int tid  = threadIdx.x;
    const int lane = tid & 31;
    const int wave = tid >> 5;
    const int mbase = blockIdx.y * BM;
    const int nbase = blockIdx.x * BN;
    const int qRow = (wave >> 1) * (BM / 2);
    const int qCol = (wave & 1) * (BN / 2);
    const int nk = (K + 63) / 64;                   // stages of K=64
    constexpr int WAITN = 2 * (BM / 32 + BN / 32);  // async instrs per stage

    v8f acc[TM][TN];
    #pragma unroll
    for (int i = 0; i < TM; i++)
        #pragma unroll
        for (int j = 0; j < TN; j++) { v8f z = {}; acc[i][j] = z; }

    auto stage = [&](int s, int buf) {
        #pragma unroll
        for (int c = 0; c < 2; c++) {
            int kb = s * 64 + c * 32;
            if (kb < K) {
                #pragma unroll
                for (int i = 0; i < BM / 32; i++) {
                    int idx = tid + i * 128;            // BM rows x 4 x 16B
                    int row = idx >> 2, seg = idx & 3;
                    const unsigned short* gp =
                        A + (size_t)(mbase + row) * K + kb + seg * 8;
                    unsigned laddr = (unsigned)(unsigned long long)(uintptr_t)
                        &As[buf][c][row * LDT + seg * 8];
                    asm volatile("global_load_async_to_lds_b128 %0, %1, off"
                                 :: "v"(laddr), "v"(gp) : "memory");
                }
                #pragma unroll
                for (int i = 0; i < BN / 32; i++) {
                    int idx = tid + i * 128;
                    int row = idx >> 2, seg = idx & 3;
                    int n = nbase + row;
                    if (n >= N) n = N - 1;              // cols >= N never stored
                    const unsigned short* gp =
                        Bw + (size_t)n * K + kb + seg * 8;
                    unsigned laddr = (unsigned)(unsigned long long)(uintptr_t)
                        &Bs[buf][c][row * LDT + seg * 8];
                    asm volatile("global_load_async_to_lds_b128 %0, %1, off"
                                 :: "v"(laddr), "v"(gp) : "memory");
                }
            }
        }
    };

    stage(0, 0);
    for (int s = 0; s < nk; s++) {
        int buf = s & 1;
        if (s + 1 < nk) {
            stage(s + 1, buf ^ 1);
            // wait for stage s (leave stage s+1 in flight)
            asm volatile("s_wait_asynccnt %0" :: "n"(WAITN) : "memory");
        } else {
            asm volatile("s_wait_asynccnt 0x0" ::: "memory");
        }
        __syncthreads();

        union Frag { uint4 q[2]; bf16x16 v; };
        #pragma unroll
        for (int c = 0; c < 2; c++) {
            if (s * 64 + c * 32 < K) {
                Frag fa[TM], fb[TN];
                #pragma unroll
                for (int mt = 0; mt < TM; mt++) {
                    int m = qRow + mt * 16 + (lane & 15);
                    int koff = (lane >> 4) * 8;
                    fa[mt].q[0] = *reinterpret_cast<const uint4*>(&As[buf][c][m * LDT + koff]);
                    fa[mt].q[1] = *reinterpret_cast<const uint4*>(&As[buf][c][m * LDT + 16 + koff]);
                }
                #pragma unroll
                for (int nt = 0; nt < TN; nt++) {
                    int n = qCol + nt * 16 + (lane & 15);
                    int k0 = (lane >> 4) * 8;
                    fb[nt].q[0] = *reinterpret_cast<const uint4*>(&Bs[buf][c][n * LDT + k0]);
                    fb[nt].q[1] = *reinterpret_cast<const uint4*>(&Bs[buf][c][n * LDT + 16 + k0]);
                }
                #pragma unroll
                for (int mt = 0; mt < TM; mt++)
                    #pragma unroll
                    for (int nt = 0; nt < TN; nt++)
                        acc[mt][nt] = __builtin_amdgcn_wmma_f32_16x16x32_bf16(
                            false, fa[mt].v, false, fb[nt].v,
                            (short)0, acc[mt][nt], false, false);
            }
        }
        __syncthreads();
    }

    // ---- epilogue: optional activation, fp32 and/or bf16 store ----
    #pragma unroll
    for (int mt = 0; mt < TM; mt++)
        #pragma unroll
        for (int nt = 0; nt < TN; nt++) {
            int col = nbase + qCol + nt * 16 + (lane & 15);
            if (col >= N) continue;
            int row0 = mbase + qRow + mt * 16 + (lane >> 4) * 8;
            #pragma unroll
            for (int j = 0; j < 8; j++) {
                float vl = acc[mt][nt][j];
                if (act == 1)      vl = tanhf(vl);
                else if (act == 2) vl = sigmoidf_(vl);
                size_t o = (size_t)(row0 + j) * N + col;
                if (outF) outF[o] = vl;
                if (outB) outB[o] = f2bf(vl);
            }
        }
}

// ---------------------------------------------------------------------------
// Fuse: decay w -> exp(w), value residual mix, in-context a-gate, kk L2 norm.
// In-place: zw->exp(w), za->aa=-kk, zv->bb=kk*a; k,v updated in place.
// ---------------------------------------------------------------------------
__global__ void __launch_bounds__(256)
fuse_kernel(const float* __restrict__ v_first,
            const float* __restrict__ w0, const float* __restrict__ a0,
            const float* __restrict__ v0, const float* __restrict__ k_k,
            const float* __restrict__ k_a,
            float* __restrict__ kbuf, float* __restrict__ vbuf,
            float* __restrict__ zw_ew, float* __restrict__ za_aa,
            float* __restrict__ zv_bb)
{
    __shared__ float s_red[256];
    int t = blockIdx.x;
    int tid = threadIdx.x;
    int c0 = tid * 4;
    size_t base = (size_t)t * C_DIM;

    float kv[4], kkr[4], ss = 0.f;
    #pragma unroll
    for (int q = 0; q < 4; q++) {
        int c = c0 + q;
        kv[q]  = kbuf[base + c];
        kkr[q] = kv[q] * k_k[c];
        ss += kkr[q] * kkr[q];
    }
    s_red[tid] = ss;
    __syncthreads();
    int gb = tid & ~15;
    float tot = 0.f;
    #pragma unroll
    for (int i = 0; i < 16; i++) tot += s_red[gb + i];
    float inv = 1.f / fmaxf(sqrtf(tot), 1e-12f);

    #pragma unroll
    for (int q = 0; q < 4; q++) {
        int c = c0 + q;
        size_t idx = base + c;
        float z  = w0[c] + zw_ew[idx];
        float sp = log1pf(__expf(-z));              // softplus(-z)
        float w  = -__expf(-sp - 0.5f);
        zw_ew[idx] = __expf(w);
        float a  = sigmoidf_(a0[c] + za_aa[idx]);
        float sv = sigmoidf_(v0[c] + zv_bb[idx]);
        float vv = vbuf[idx];
        vbuf[idx] = vv + (v_first[idx] - vv) * sv;
        float kkn = kkr[q] * inv;
        za_aa[idx] = -kkn;          // aa
        zv_bb[idx] = kkn * a;       // bb
        kbuf[idx]  = kv[q] * (1.f + (a - 1.f) * k_a[c]);
    }
}

// ---------------------------------------------------------------------------
// Sequential selective-decay scan: S = S*diag(ew) + (S a)b^T + v k^T ; o = S r
// Grid = H*4 (value rows split 4-way/head); 256 thr = 16 rows x 16 lanes.
// Software-pipelined: step t+1's vectors prefetched into registers while
// step t computes (global latency hidden behind the recurrence).
// ---------------------------------------------------------------------------
__global__ void __launch_bounds__(256)
scan_kernel(const float* __restrict__ r, const float* __restrict__ ew,
            const float* __restrict__ k, const float* __restrict__ v,
            const float* __restrict__ aa, const float* __restrict__ bb,
            const float* __restrict__ init_state, float* __restrict__ y)
{
    const int h     = blockIdx.x >> 2;
    const int vbase = (blockIdx.x & 3) * 16;
    const int tid   = threadIdx.x;
    const int vr    = vbase + (tid >> 4);
    const int l16   = tid & 15;
    const int j0    = l16 * 4;

    float S[4];
    #pragma unroll
    for (int q = 0; q < 4; q++)
        S[q] = init_state[((size_t)h * HS_ + vr) * HS_ + j0 + q];

    __shared__ float sr[64], sw[64], sk[64], sv[64], sa[64], sb[64];

    float pA = 0.f, pB = 0.f;
    auto preload = [&](int t) {
        size_t base = (size_t)t * C_DIM + h * HS_;
        if (tid < 64)       { pA = r[base + tid];        pB = ew[base + tid]; }
        else if (tid < 128) { pA = k[base + tid - 64];   pB = v[base + tid - 64]; }
        else if (tid < 192) { pA = aa[base + tid - 128]; pB = bb[base + tid - 128]; }
    };
    preload(0);

    for (int t = 0; t < T_LEN; t++) {
        if (tid < 64)       { sr[tid]       = pA; sw[tid]       = pB; }
        else if (tid < 128) { sk[tid - 64]  = pA; sv[tid - 64]  = pB; }
        else if (tid < 192) { sa[tid - 128] = pA; sb[tid - 128] = pB; }
        __syncthreads();
        if (t + 1 < T_LEN) preload(t + 1);      // overlap with compute below

        float vv  = sv[vr];
        float sap = S[0]*sa[j0] + S[1]*sa[j0+1] + S[2]*sa[j0+2] + S[3]*sa[j0+3];
        sap += __shfl_xor(sap, 1, 32);
        sap += __shfl_xor(sap, 2, 32);
        sap += __shfl_xor(sap, 4, 32);
        sap += __shfl_xor(sap, 8, 32);

        float o = 0.f;
        #pragma unroll
        for (int q = 0; q < 4; q++) {
            int j = j0 + q;
            float s = S[q] * sw[j] + sap * sb[j] + vv * sk[j];
            S[q] = s;
            o += s * sr[j];
        }
        o += __shfl_xor(o, 1, 32);
        o += __shfl_xor(o, 2, 32);
        o += __shfl_xor(o, 4, 32);
        o += __shfl_xor(o, 8, 32);
        size_t base = (size_t)t * C_DIM + h * HS_;
        if (l16 == 0) y[base + vr] = o;
        __syncthreads();
    }
}

// ---------------------------------------------------------------------------
// GroupNorm(H groups, eps=6.4e-4) + per-head bonus + output gate -> bf16.
// ---------------------------------------------------------------------------
__global__ void __launch_bounds__(256)
gn_bonus_gate_kernel(const float* __restrict__ y, const float* __restrict__ r,
                     const float* __restrict__ k, const float* __restrict__ v,
                     const float* __restrict__ g, const float* __restrict__ r_k,
                     const float* __restrict__ ln_w, const float* __restrict__ ln_b,
                     unsigned short* __restrict__ xog)
{
    __shared__ float s1[256], s2[256], s3[256];
    int t = blockIdx.x;
    int tid = threadIdx.x;
    int c0 = tid * 4;
    size_t base = (size_t)t * C_DIM;

    float yv[4], p1 = 0.f, p2 = 0.f, p3 = 0.f;
    #pragma unroll
    for (int q = 0; q < 4; q++) {
        int c = c0 + q;
        yv[q] = y[base + c];
        p1 += yv[q];
        p2 += yv[q] * yv[q];
        p3 += r[base + c] * k[base + c] * r_k[c];
    }
    s1[tid] = p1; s2[tid] = p2; s3[tid] = p3;
    __syncthreads();
    int gb = tid & ~15;
    float S1 = 0.f, S2 = 0.f, S3 = 0.f;
    #pragma unroll
    for (int i = 0; i < 16; i++) { S1 += s1[gb+i]; S2 += s2[gb+i]; S3 += s3[gb+i]; }
    float mu  = S1 * (1.f / 64.f);
    float var = S2 * (1.f / 64.f) - mu * mu;
    float inv = rsqrtf(var + EPS_GN);

    #pragma unroll
    for (int q = 0; q < 4; q++) {
        int c = c0 + q;
        float yn = (yv[q] - mu) * inv * ln_w[c] + ln_b[c];
        float xo = yn + S3 * v[base + c];
        xog[base + c] = f2bf(xo * g[base + c]);
    }
}

// ---------------------------------------------------------------------------
extern "C" void kernel_launch(void* const* d_in, const int* in_sizes, int n_in,
                              void* d_out, int out_size, void* d_ws, size_t ws_size,
                              hipStream_t stream)
{
    (void)in_sizes; (void)n_in; (void)out_size; (void)ws_size;
    const float* x        = (const float*)d_in[0];
    const float* v_first  = (const float*)d_in[1];
    const float* x_prev   = (const float*)d_in[2];
    const float* init_st  = (const float*)d_in[3];
    const float* x_r = (const float*)d_in[4];
    const float* x_w = (const float*)d_in[5];
    const float* x_k = (const float*)d_in[6];
    const float* x_v = (const float*)d_in[7];
    const float* x_a = (const float*)d_in[8];
    const float* x_g = (const float*)d_in[9];
    const float* w0  = (const float*)d_in[10];
    const float* a0  = (const float*)d_in[11];
    const float* v0  = (const float*)d_in[12];
    const float* k_k = (const float*)d_in[13];
    const float* k_a = (const float*)d_in[14];
    const float* w1  = (const float*)d_in[15];
    const float* w2  = (const float*)d_in[16];
    const float* a1  = (const float*)d_in[17];
    const float* a2  = (const float*)d_in[18];
    const float* v1  = (const float*)d_in[19];
    const float* v2  = (const float*)d_in[20];
    const float* g1  = (const float*)d_in[21];
    const float* g2  = (const float*)d_in[22];
    const float* r_k = (const float*)d_in[23];
    const float* Wr  = (const float*)d_in[24];
    const float* Wk  = (const float*)d_in[25];
    const float* Wv  = (const float*)d_in[26];
    const float* Wo  = (const float*)d_in[27];
    const float* lnw = (const float*)d_in[28];
    const float* lnb = (const float*)d_in[29];
    // d_in[30] = cu_seqlens: single sequence [0, T]; start handled as t==0.

    const size_t TC = (size_t)T_LEN * C_DIM;
    char* p = (char*)d_ws;
    auto alloc = [&](size_t bytes) -> char* {
        char* q = p; p += (bytes + 255) & ~(size_t)255; return q;
    };
    // bf16 activations
    unsigned short* xr  = (unsigned short*)alloc(TC * 2);
    unsigned short* xw  = (unsigned short*)alloc(TC * 2);
    unsigned short* xk  = (unsigned short*)alloc(TC * 2);
    unsigned short* xv  = (unsigned short*)alloc(TC * 2);
    unsigned short* xa  = (unsigned short*)alloc(TC * 2);
    unsigned short* xg  = (unsigned short*)alloc(TC * 2);
    unsigned short* hw  = (unsigned short*)alloc((size_t)T_LEN * 64 * 2);
    unsigned short* ha  = (unsigned short*)alloc((size_t)T_LEN * 64 * 2);
    unsigned short* hv  = (unsigned short*)alloc((size_t)T_LEN * 32 * 2);
    unsigned short* hg  = (unsigned short*)alloc((size_t)T_LEN * 128 * 2);
    unsigned short* xog = (unsigned short*)alloc(TC * 2);
    // bf16 canonical [N,K] weights
    unsigned short* cWr = (unsigned short*)alloc((size_t)C_DIM * C_DIM * 2);
    unsigned short* cWk = (unsigned short*)alloc((size_t)C_DIM * C_DIM * 2);
    unsigned short* cWv = (unsigned short*)alloc((size_t)C_DIM * C_DIM * 2);
    unsigned short* cWo = (unsigned short*)alloc((size_t)C_DIM * C_DIM * 2);
    unsigned short* cw1 = (unsigned short*)alloc((size_t)64  * C_DIM * 2);
    unsigned short* cw2 = (unsigned short*)alloc((size_t)C_DIM * 64  * 2);
    unsigned short* ca1 = (unsigned short*)alloc((size_t)64  * C_DIM * 2);
    unsigned short* ca2 = (unsigned short*)alloc((size_t)C_DIM * 64  * 2);
    unsigned short* cv1 = (unsigned short*)alloc((size_t)32  * C_DIM * 2);
    unsigned short* cv2 = (unsigned short*)alloc((size_t)C_DIM * 32  * 2);
    unsigned short* cg1 = (unsigned short*)alloc((size_t)128 * C_DIM * 2);
    unsigned short* cg2 = (unsigned short*)alloc((size_t)C_DIM * 128 * 2);
    // fp32 activations
    float* rb   = (float*)alloc(TC * 4);
    float* kb   = (float*)alloc(TC * 4);
    float* vb   = (float*)alloc(TC * 4);
    float* ewb  = (float*)alloc(TC * 4);   // zw -> exp(w)
    float* aab  = (float*)alloc(TC * 4);   // za -> aa
    float* bbb  = (float*)alloc(TC * 4);   // zv -> bb
    float* gbuf = (float*)alloc(TC * 4);
    float* yb   = (float*)alloc(TC * 4);

    dim3 b256(256), b128(128);

    auto conv = [&](const float* src, unsigned short* dst, int K, int N, int srcIsNK) {
        size_t total = (size_t)N * K;
        convert_weight_kernel<<<dim3((unsigned)((total + 255) / 256)), b256, 0, stream>>>(
            src, dst, K, N, srcIsNK);
    };
    conv(Wr, cWr, C_DIM, C_DIM, 1);
    conv(Wk, cWk, C_DIM, C_DIM, 1);
    conv(Wv, cWv, C_DIM, C_DIM, 1);
    conv(Wo, cWo, C_DIM, C_DIM, 1);
    conv(w1, cw1, C_DIM, 64,  0);   // [K=1024,N=64]   -> [64,1024]
    conv(w2, cw2, 64,  C_DIM, 0);   // [K=64,N=1024]   -> [1024,64]
    conv(a1, ca1, C_DIM, 64,  0);
    conv(a2, ca2, 64,  C_DIM, 0);
    conv(v1, cv1, C_DIM, 32,  0);
    conv(v2, cv2, 32,  C_DIM, 0);
    conv(g1, cg1, C_DIM, 128, 0);
    conv(g2, cg2, 128, C_DIM, 0);

    mix_kernel<<<T_LEN, b256, 0, stream>>>(x, x_prev, x_r, x_w, x_k, x_v, x_a, x_g,
                                           xr, xw, xk, xv, xa, xg);

    // Large-N GEMMs: 128x64 block tile (TM=4, TN=2) -> 16 WMMA / 12 frag loads.
    auto gemmL = [&](const unsigned short* A, const unsigned short* B, int N, int K,
                     int act, float* oF, unsigned short* oB) {
        dim3 grid((N + 63) / 64, T_LEN / 128);
        gemm_bf16_wmma<128, 64, 4, 2><<<grid, b128, 0, stream>>>(
            A, B, T_LEN, N, K, act, oF, oB);
    };
    // Small-N GEMMs: 64x64 block tile for M-parallelism / occupancy.
    auto gemmS = [&](const unsigned short* A, const unsigned short* B, int N, int K,
                     int act, float* oF, unsigned short* oB) {
        dim3 grid((N + 63) / 64, T_LEN / 64);
        gemm_bf16_wmma<64, 64, 2, 2><<<grid, b128, 0, stream>>>(
            A, B, T_LEN, N, K, act, oF, oB);
    };

    gemmL(xr, cWr, C_DIM, C_DIM, 0, rb, nullptr);          // r
    gemmL(xk, cWk, C_DIM, C_DIM, 0, kb, nullptr);          // k
    gemmL(xv, cWv, C_DIM, C_DIM, 0, vb, nullptr);          // v
    gemmS(xw, cw1, 64,    C_DIM, 1, nullptr, hw);          // tanh(xw@w1)
    gemmL(hw, cw2, C_DIM, 64,    0, ewb, nullptr);         // zw
    gemmS(xa, ca1, 64,    C_DIM, 0, nullptr, ha);          // xa@a1
    gemmL(ha, ca2, C_DIM, 64,    0, aab, nullptr);         // za
    gemmS(xv, cv1, 32,    C_DIM, 0, nullptr, hv);          // xv@v1
    gemmL(hv, cv2, C_DIM, 32,    0, bbb, nullptr);         // zv
    gemmS(xg, cg1, 128,   C_DIM, 2, nullptr, hg);          // sigmoid(xg@g1)
    gemmL(hg, cg2, C_DIM, 128,   0, gbuf, nullptr);        // g

    fuse_kernel<<<T_LEN, b256, 0, stream>>>(v_first, w0, a0, v0, k_k, k_a,
                                            kb, vb, ewb, aab, bbb);

    scan_kernel<<<H_N * 4, b256, 0, stream>>>(rb, ewb, kb, vb, aab, bbb, init_st, yb);

    gn_bonus_gate_kernel<<<T_LEN, b256, 0, stream>>>(yb, rb, kb, vb, gbuf, r_k,
                                                     lnw, lnb, xog);

    gemmL(xog, cWo, C_DIM, C_DIM, 0, (float*)d_out, nullptr);   // output proj
}
